// EncoderLayer_82867099009759
// MI455X (gfx1250) — matrix operand order repaired and checked by reading, not compiled, not run on hardware
//
#include <hip/hip_runtime.h>
#include <hip/hip_bf16.h>
#include <math.h>

// ---------------------------------------------------------------- types ----
typedef __attribute__((ext_vector_type(16))) __bf16 v16bf;
typedef __attribute__((ext_vector_type(8)))  float  v8f;

#define BDIM   768
#define SEQ    2048
#define BATCH  2
#define HEADS  12
#define DKH    64
#define DFFDIM 3072
#define ROWS   (BATCH * SEQ)   /* 4096 */
#define LN_EPS 1e-6f
#define NEGV   -1e9f

__device__ __forceinline__ __bf16 f2bf(float f) {
  union { float f; unsigned u; } c; c.f = f;
  unsigned r = c.u + 0x7FFFu + ((c.u >> 16) & 1u);
  unsigned short h = (unsigned short)(r >> 16);
  return __builtin_bit_cast(__bf16, h);
}
__device__ __forceinline__ unsigned pack2bf(float a, float b) {
  unsigned lo = (unsigned)__builtin_bit_cast(unsigned short, f2bf(a));
  unsigned hi = (unsigned)__builtin_bit_cast(unsigned short, f2bf(b));
  return lo | (hi << 16);
}

// ------------------------------------------------------------- converts ----
__global__ __launch_bounds__(256) void f32_to_bf16_kernel(
    const float* __restrict__ in, __bf16* __restrict__ out, int n) {
  int i = blockIdx.x * blockDim.x + threadIdx.x;
  int stride = gridDim.x * blockDim.x;
  for (; i < n; i += stride) out[i] = f2bf(in[i]);
}

// in[k][n] (Kd x Nd, f32) -> out[n][k] (Nd x Kd, bf16)
__global__ __launch_bounds__(256) void f32_to_bf16_t_kernel(
    const float* __restrict__ in, __bf16* __restrict__ out, int Kd, int Nd) {
  int i = blockIdx.x * blockDim.x + threadIdx.x;
  int stride = gridDim.x * blockDim.x;
  int n_tot = Kd * Nd;
  for (; i < n_tot; i += stride) {
    int k = i / Nd, n = i - k * Nd;
    out[(size_t)n * Kd + k] = f2bf(in[i]);
  }
}

// ------------------------------------------------------- bf16 WMMA GEMM ----
// C[M,N] = A[M,K] @ Bt[N,K]^T + bias, optional ReLU. bf16 in, f32 accum.
// 256 thr = 8 waves (4 in M x 2 in N). Block tile 128x128, K-step 32.
// Wave tile 32x64: 2 A-frags x 4 B-frags -> 8 WMMA per K-step.
// Double-buffered LDS, ONE barrier per K-step.
// OUTMODE: 0 = f32 row-major, 1 = bf16 row-major, 2 = bf16 [B,H,DK,S]
#define LDT 40   // 80-byte padded rows: 16B-aligned b128 stores, conflict-free
template<int OUTMODE, bool RELU>
__global__ __launch_bounds__(256) void gemm_bf16_wmma_kernel(
    const __bf16* __restrict__ A,   // [M,K]
    const __bf16* __restrict__ Bt,  // [N,K] pre-transposed
    const float* __restrict__ bias,
    void* __restrict__ Cout,
    int M, int N, int K)
{
  __shared__ __bf16 As[2][128 * LDT];  // As[buf][m][k]
  __shared__ __bf16 Bs[2][128 * LDT];  // Bs[buf][n][k]

  const int tid  = threadIdx.x;
  const int lane = tid & 31;
  const int wave = tid >> 5;
  const int wm = wave >> 1, wn = wave & 1;
  const int m0 = blockIdx.y * 128;
  const int n0 = blockIdx.x * 128;
  const int mrow = lane & 15;
  const int half = lane >> 4;
  const int nlane = lane & 15;

  v8f acc[2][4];
  #pragma unroll
  for (int t = 0; t < 2; ++t)
    #pragma unroll
    for (int u = 0; u < 4; ++u) acc[t][u] = (v8f){};

  // staging map: thread -> (row 0..127, 16 contiguous k)
  const int sr = tid >> 1;
  const int sc = (tid & 1) * 16;
  const __bf16* Aptr = A  + (size_t)(m0 + sr) * K + sc;
  const __bf16* Bptr = Bt + (size_t)(n0 + sr) * K + sc;

  const int kIters = K >> 5;

  // preload tile 0
  uint4 avr0, avr1, bvr0, bvr1;
  {
    const uint4* a4 = (const uint4*)Aptr;
    const uint4* b4 = (const uint4*)Bptr;
    avr0 = a4[0]; avr1 = a4[1];
    bvr0 = b4[0]; bvr1 = b4[1];
  }

  for (int it = 0; it < kIters; ++it) {
    const int cur = it & 1;
    // commit tile(it) to LDS
    {
      uint4* asd = (uint4*)&As[cur][sr * LDT + sc];
      uint4* bsd = (uint4*)&Bs[cur][sr * LDT + sc];
      asd[0] = avr0; asd[1] = avr1;
      bsd[0] = bvr0; bsd[1] = bvr1;
    }
    __syncthreads();
    // fetch tile(it+1) into regs (overlaps the WMMAs below)
    if (it + 1 < kIters) {
      const int k0 = (it + 1) << 5;
      __builtin_prefetch(Aptr + k0 + 32, 0, 1);
      __builtin_prefetch(Bptr + k0 + 32, 0, 1);
      const uint4* a4 = (const uint4*)(Aptr + k0);
      const uint4* b4 = (const uint4*)(Bptr + k0);
      avr0 = a4[0]; avr1 = a4[1];
      bvr0 = b4[0]; bvr1 = b4[1];
    }

    v16bf af[2], bf[4];
    #pragma unroll
    for (int j = 0; j < 8; ++j) {
      // A-operand: lanes 0-15 K=0-15 ; lanes 16-31 K=16-31
      int ka = half * 16 + 2 * j;
      // B-operand: lanes 0-15 K=0-15 ; lanes 16-31 K=16-31
      int kb = half * 16 + 2 * j;
      #pragma unroll
      for (int t = 0; t < 2; ++t) {
        int arow = wm * 32 + t * 16 + mrow;
        af[t][2 * j]     = As[cur][arow * LDT + ka];
        af[t][2 * j + 1] = As[cur][arow * LDT + ka + 1];
      }
      #pragma unroll
      for (int u = 0; u < 4; ++u) {
        int brow = wn * 64 + u * 16 + mrow;
        bf[u][2 * j]     = Bs[cur][brow * LDT + kb];
        bf[u][2 * j + 1] = Bs[cur][brow * LDT + kb + 1];
      }
    }
    #pragma unroll
    for (int t = 0; t < 2; ++t)
      #pragma unroll
      for (int u = 0; u < 4; ++u)
        acc[t][u] = __builtin_amdgcn_wmma_f32_16x16x32_bf16(
            false, af[t], false, bf[u], (short)0, acc[t][u], false, false);
  }

  // epilogue: C layout VGPR r -> M = r + 8*half, N = lane&15
  float bv[4];
  #pragma unroll
  for (int u = 0; u < 4; ++u) bv[u] = bias[n0 + wn * 64 + u * 16 + nlane];

  #pragma unroll
  for (int t = 0; t < 2; ++t) {
    #pragma unroll
    for (int r = 0; r < 8; ++r) {
      const int row = m0 + wm * 32 + t * 16 + r + half * 8;
      #pragma unroll
      for (int u = 0; u < 4; ++u) {
        const int col = n0 + wn * 64 + u * 16 + nlane;
        float v = acc[t][u][r] + bv[u];
        if (RELU) v = fmaxf(v, 0.f);
        if (OUTMODE == 0) {
          ((float*)Cout)[(size_t)row * N + col] = v;
        } else if (OUTMODE == 1) {
          ((__bf16*)Cout)[(size_t)row * N + col] = f2bf(v);
        } else {  // [B,H,DK,S] d-major per head (for attention V)
          int bb = row >> 11, ss = row & 2047;
          int hh = col >> 6,  dd = col & 63;
          ((__bf16*)Cout)[(((size_t)bb * HEADS + hh) * DKH + dd) * SEQ + ss] =
              f2bf(v);
        }
      }
    }
  }
}

// --------------------------------------------------- flash attention -------
// One block per (b,h, 64-query rows). 8 waves (4xM, 2xN). Online softmax.
// Mask is applied in the row-owned softmax phase where it is contiguous
// (uint4 loads), not in the WMMA-lane domain.
#define APAD 72   // 144-byte bf16 rows: 16B-aligned b128 stores, conflict-free
#define SPAD 68   // 272-byte f32 rows: 16B-aligned float4 access
__global__ __launch_bounds__(256) void attention_kernel(
    const __bf16* __restrict__ Q,   // [B*S, D]
    const __bf16* __restrict__ K,   // [B*S, D]
    const __bf16* __restrict__ Vt,  // [B,H,DK,S]
    const int* __restrict__ mask,
    __bf16* __restrict__ ctx)       // [B*S, D]
{
  __shared__ __bf16 Qs[64 * APAD];  // Qs[q][d]
  __shared__ __bf16 Ks[64 * APAD];  // Ks[key][d] == B^T form for Q.K^T
  __shared__ __bf16 Vs[64 * APAD];  // Vs[d][key] == B^T form for P.V
  __shared__ __bf16 Ps[64 * APAD];  // probabilities (bf16)
  __shared__ float  Sf[64 * SPAD];  // raw scores (f32)
  __shared__ float  m_i[64], l_i[64], sc_i[64];

  const int tid  = threadIdx.x;
  const int lane = tid & 31, wave = tid >> 5;
  const int wm = wave >> 1, wn = wave & 1;
  const int mrow = lane & 15, half = lane >> 4, nlane = lane & 15;

  const int bh = blockIdx.y;
  const int b = bh / HEADS, h = bh % HEADS;
  const int q0 = blockIdx.x * 64;
  const size_t base  = (size_t)b * SEQ * BDIM + (size_t)h * DKH;
  const size_t vbase = ((size_t)bh * DKH) * SEQ;

  const int sr = tid >> 2;          // row 0..63
  const int sc = (tid & 3) * 16;    // 16 contiguous elems

  // stage Q block (b128 copies)
  {
    const uint4* src = (const uint4*)(Q + base + (size_t)(q0 + sr) * BDIM + sc);
    uint4* dst = (uint4*)&Qs[sr * APAD + sc];
    dst[0] = src[0]; dst[1] = src[1];
  }
  if (tid < 64) { m_i[tid] = -INFINITY; l_i[tid] = 0.f; }
  __syncthreads();

  v8f o0 = {}; v8f o1 = {};

  for (int j0 = 0; j0 < SEQ; j0 += 64) {
    // stage K (row-major) and V (already d-major in memory) with b128 copies
    {
      const uint4* ks = (const uint4*)(K + base + (size_t)(j0 + sr) * BDIM + sc);
      const uint4* vs = (const uint4*)(Vt + vbase + (size_t)sr * SEQ + j0 + sc);
      uint4* kd = (uint4*)&Ks[sr * APAD + sc];
      uint4* vd = (uint4*)&Vs[sr * APAD + sc];
      kd[0] = ks[0]; kd[1] = ks[1];
      vd[0] = vs[0]; vd[1] = vs[1];
    }
    __syncthreads();

    // S = Q @ K^T (d = 64 -> two K-steps of 32)
    v8f s0 = {}; v8f s1 = {};
    #pragma unroll
    for (int kk = 0; kk < 64; kk += 32) {
      v16bf a, b0, b1;
      #pragma unroll
      for (int j = 0; j < 8; ++j) {
        int ka   = kk + half * 16 + 2 * j;
        int arow = wm * 16 + mrow;
        a[2 * j]     = Qs[arow * APAD + ka];
        a[2 * j + 1] = Qs[arow * APAD + ka + 1];
        int kb  = kk + half * 16 + 2 * j;
        int br0 = wn * 32 + mrow, br1 = br0 + 16;
        b0[2 * j]     = Ks[br0 * APAD + kb];
        b0[2 * j + 1] = Ks[br0 * APAD + kb + 1];
        b1[2 * j]     = Ks[br1 * APAD + kb];
        b1[2 * j + 1] = Ks[br1 * APAD + kb + 1];
      }
      s0 = __builtin_amdgcn_wmma_f32_16x16x32_bf16(false, a, false, b0,
                                                   (short)0, s0, false, false);
      s1 = __builtin_amdgcn_wmma_f32_16x16x32_bf16(false, a, false, b1,
                                                   (short)0, s1, false, false);
    }

    // scale 1/sqrt(64) and spill (mask handled in softmax phase)
    #pragma unroll
    for (int r = 0; r < 8; ++r) {
      int lr  = wm * 16 + r + half * 8;
      int lc0 = wn * 32 + nlane, lc1 = lc0 + 16;
      Sf[lr * SPAD + lc0] = s0[r] * 0.125f;
      Sf[lr * SPAD + lc1] = s1[r] * 0.125f;
    }
    __syncthreads();

    // online softmax row update (one thread per query row).
    // Mask row is contiguous here -> uint4 loads; scores via float4.
    if (tid < 64) {
      const uint4* mrow4 =
          (const uint4*)(mask + (size_t)(q0 + tid) * SEQ + j0);
      float* srow = &Sf[tid * SPAD];
      float4* srow4 = (float4*)srow;
      float rmax = -INFINITY;
      #pragma unroll 4
      for (int c4 = 0; c4 < 16; ++c4) {
        uint4 mv = mrow4[c4];
        float4 sv = srow4[c4];
        sv.x = mv.x ? sv.x : NEGV;
        sv.y = mv.y ? sv.y : NEGV;
        sv.z = mv.z ? sv.z : NEGV;
        sv.w = mv.w ? sv.w : NEGV;
        srow4[c4] = sv;
        rmax = fmaxf(rmax, fmaxf(fmaxf(sv.x, sv.y), fmaxf(sv.z, sv.w)));
      }
      float mold = m_i[tid];
      float mnew = fmaxf(mold, rmax);
      float scale = __expf(mold - mnew);        // 0 on first iteration
      float rsum = 0.f;
      uint32_t* prow = (uint32_t*)&Ps[tid * APAD];
      #pragma unroll 4
      for (int c4 = 0; c4 < 16; ++c4) {
        float4 sv = srow4[c4];
        float p0 = __expf(sv.x - mnew);
        float p1 = __expf(sv.y - mnew);
        float p2 = __expf(sv.z - mnew);
        float p3 = __expf(sv.w - mnew);
        prow[2 * c4]     = pack2bf(p0, p1);
        prow[2 * c4 + 1] = pack2bf(p2, p3);
        rsum += (p0 + p1) + (p2 + p3);
      }
      l_i[tid]  = l_i[tid] * scale + rsum;
      m_i[tid]  = mnew;
      sc_i[tid] = scale;
    }
    __syncthreads();

    // rescale O, then O += P @ V
    #pragma unroll
    for (int r = 0; r < 8; ++r) {
      float f = sc_i[wm * 16 + r + half * 8];
      o0[r] *= f; o1[r] *= f;
    }
    #pragma unroll
    for (int kk = 0; kk < 64; kk += 32) {
      v16bf a, b0, b1;
      #pragma unroll
      for (int j = 0; j < 8; ++j) {
        int ka   = kk + half * 16 + 2 * j;
        int arow = wm * 16 + mrow;
        a[2 * j]     = Ps[arow * APAD + ka];
        a[2 * j + 1] = Ps[arow * APAD + ka + 1];
        int kb  = kk + half * 16 + 2 * j;
        int br0 = wn * 32 + mrow, br1 = br0 + 16;   // n == head-dim index
        b0[2 * j]     = Vs[br0 * APAD + kb];
        b0[2 * j + 1] = Vs[br0 * APAD + kb + 1];
        b1[2 * j]     = Vs[br1 * APAD + kb];
        b1[2 * j + 1] = Vs[br1 * APAD + kb + 1];
      }
      o0 = __builtin_amdgcn_wmma_f32_16x16x32_bf16(false, a, false, b0,
                                                   (short)0, o0, false, false);
      o1 = __builtin_amdgcn_wmma_f32_16x16x32_bf16(false, a, false, b1,
                                                   (short)0, o1, false, false);
    }
    __syncthreads();
  }

  // finalize: O /= l ; write ctx (bf16, [B*S, D] with head offset)
  #pragma unroll
  for (int r = 0; r < 8; ++r) {
    int lr  = wm * 16 + r + half * 8;
    float inv = 1.f / l_i[lr];
    int qg = q0 + lr;
    int d0 = wn * 32 + nlane, d1 = d0 + 16;
    ctx[base + (size_t)qg * BDIM + d0] = f2bf(o0[r] * inv);
    ctx[base + (size_t)qg * BDIM + d1] = f2bf(o1[r] * inv);
  }
}

// ------------------------------------------- residual + layernorm ----------
// torch semantics: mean, UNBIASED std (ddof=1), eps added to std.
template<bool WRITE_BF16>
__global__ __launch_bounds__(256) void add_layernorm_kernel(
    const float* __restrict__ xa, const float* __restrict__ xb,
    const float* __restrict__ alpha, const float* __restrict__ beta,
    float* __restrict__ outf, __bf16* __restrict__ outbf)
{
  __shared__ float red[256];
  __shared__ float red2[256];
  __shared__ float s_mean, s_inv;
  const int row = blockIdx.x;
  const int tid = threadIdx.x;
  const float* pa = xa + (size_t)row * BDIM;
  const float* pb = xb + (size_t)row * BDIM;

  float v[3];
  float s = 0.f, sq = 0.f;
  #pragma unroll
  for (int i = 0; i < 3; ++i) {
    float t = pa[tid + i * 256] + pb[tid + i * 256];
    v[i] = t; s += t; sq += t * t;
  }
  red[tid] = s; red2[tid] = sq;
  __syncthreads();
  for (int off = 128; off > 0; off >>= 1) {
    if (tid < off) { red[tid] += red[tid + off]; red2[tid] += red2[tid + off]; }
    __syncthreads();
  }
  if (tid == 0) {
    float mean = red[0] / BDIM;
    float var  = (red2[0] - BDIM * mean * mean) / (BDIM - 1);
    var = var > 0.f ? var : 0.f;
    s_mean = mean;
    s_inv  = 1.f / (sqrtf(var) + LN_EPS);
  }
  __syncthreads();
  const float al = alpha[0], be = beta[0];
  #pragma unroll
  for (int i = 0; i < 3; ++i) {
    float y = al * (v[i] - s_mean) * s_inv + be;
    int c = tid + i * 256;
    outf[(size_t)row * BDIM + c] = y;
    if (WRITE_BF16) outbf[(size_t)row * BDIM + c] = f2bf(y);
  }
}

// ------------------------------------------------------------ launcher -----
extern "C" void kernel_launch(void* const* d_in, const int* in_sizes, int n_in,
                              void* d_out, int out_size, void* d_ws, size_t ws_size,
                              hipStream_t stream) {
  (void)in_sizes; (void)n_in; (void)out_size; (void)ws_size;
  const float* x      = (const float*)d_in[0];
  const float* wq     = (const float*)d_in[1];
  const float* bq     = (const float*)d_in[2];
  const float* wk     = (const float*)d_in[3];
  const float* bk     = (const float*)d_in[4];
  const float* wv     = (const float*)d_in[5];
  const float* bv     = (const float*)d_in[6];
  const float* wo     = (const float*)d_in[7];
  const float* bo     = (const float*)d_in[8];
  const float* w1     = (const float*)d_in[9];
  const float* b1     = (const float*)d_in[10];
  const float* w2     = (const float*)d_in[11];
  const float* b2     = (const float*)d_in[12];
  const float* alpha1 = (const float*)d_in[13];
  const float* beta1  = (const float*)d_in[14];
  const float* alpha2 = (const float*)d_in[15];
  const float* beta2  = (const float*)d_in[16];
  const int*   mask   = (const int*)d_in[17];
  float* out = (float*)d_out;

  size_t off = 0;
  auto alloc = [&](size_t bytes) -> char* {
    char* p = (char*)d_ws + off;
    off += (bytes + 255) & ~(size_t)255;
    return p;
  };

  __bf16* xb    = (__bf16*)alloc((size_t)ROWS * BDIM * 2);
  __bf16* wqbT  = (__bf16*)alloc((size_t)BDIM * BDIM * 2);
  __bf16* wkbT  = (__bf16*)alloc((size_t)BDIM * BDIM * 2);
  __bf16* wvbT  = (__bf16*)alloc((size_t)BDIM * BDIM * 2);
  __bf16* wobT  = (__bf16*)alloc((size_t)BDIM * BDIM * 2);
  __bf16* w1bT  = (__bf16*)alloc((size_t)BDIM * DFFDIM * 2);
  __bf16* w2bT  = (__bf16*)alloc((size_t)DFFDIM * BDIM * 2);
  __bf16* qb    = (__bf16*)alloc((size_t)ROWS * BDIM * 2);
  __bf16* kb    = (__bf16*)alloc((size_t)ROWS * BDIM * 2);
  __bf16* vbT   = (__bf16*)alloc((size_t)ROWS * BDIM * 2);   // [B,H,DK,S]
  __bf16* ctxb  = (__bf16*)alloc((size_t)ROWS * BDIM * 2);
  float*  attnf = (float*) alloc((size_t)ROWS * BDIM * 4);
  float*  ln1f  = (float*) alloc((size_t)ROWS * BDIM * 4);
  __bf16* ln1b  = (__bf16*)alloc((size_t)ROWS * BDIM * 2);
  // Buffer reuse (stream-ordered, producers/consumers strictly sequenced):
  __bf16* ff1b = qb;     // q/k/vT/ctx (4 x 6.29MB) dead before FFN1; 24MB need
  float*  ff2f = attnf;  // attn_out dead after LN1

  dim3 blk(256);

  // fp32 -> bf16 conversions (weights transposed to [N,K])
  f32_to_bf16_kernel  <<<1024, blk, 0, stream>>>(x,  xb,  ROWS * BDIM);
  f32_to_bf16_t_kernel<<<512,  blk, 0, stream>>>(wq, wqbT, BDIM, BDIM);
  f32_to_bf16_t_kernel<<<512,  blk, 0, stream>>>(wk, wkbT, BDIM, BDIM);
  f32_to_bf16_t_kernel<<<512,  blk, 0, stream>>>(wv, wvbT, BDIM, BDIM);
  f32_to_bf16_t_kernel<<<512,  blk, 0, stream>>>(wo, wobT, BDIM, BDIM);
  f32_to_bf16_t_kernel<<<1024, blk, 0, stream>>>(w1, w1bT, BDIM, DFFDIM);
  f32_to_bf16_t_kernel<<<1024, blk, 0, stream>>>(w2, w2bT, DFFDIM, BDIM);

  // Q,K,V projections: [4096x768] @ [768x768]
  dim3 gproj(BDIM / 128, ROWS / 128);           // (6, 32)
  gemm_bf16_wmma_kernel<1, false><<<gproj, blk, 0, stream>>>(
      xb, wqbT, bq, qb, ROWS, BDIM, BDIM);
  gemm_bf16_wmma_kernel<1, false><<<gproj, blk, 0, stream>>>(
      xb, wkbT, bk, kb, ROWS, BDIM, BDIM);
  gemm_bf16_wmma_kernel<2, false><<<gproj, blk, 0, stream>>>(
      xb, wvbT, bv, vbT, ROWS, BDIM, BDIM);

  // attention
  dim3 gattn(SEQ / 64, BATCH * HEADS);          // (32, 24)
  attention_kernel<<<gattn, blk, 0, stream>>>(qb, kb, vbT, mask, ctxb);

  // output projection
  gemm_bf16_wmma_kernel<0, false><<<gproj, blk, 0, stream>>>(
      ctxb, wobT, bo, attnf, ROWS, BDIM, BDIM);

  // LN1: x + attn_out
  add_layernorm_kernel<true><<<ROWS, blk, 0, stream>>>(
      x, attnf, alpha1, beta1, ln1f, ln1b);

  // FFN1 (+ReLU): [4096x768] @ [768x3072]
  dim3 gff1(DFFDIM / 128, ROWS / 128);          // (24, 32)
  gemm_bf16_wmma_kernel<1, true><<<gff1, blk, 0, stream>>>(
      ln1b, w1bT, b1, ff1b, ROWS, DFFDIM, BDIM);

  // FFN2: [4096x3072] @ [3072x768]
  gemm_bf16_wmma_kernel<0, false><<<gproj, blk, 0, stream>>>(
      ff1b, w2bT, b2, ff2f, ROWS, BDIM, DFFDIM);

  // LN2: ln1 + ff -> final output (f32)
  add_layernorm_kernel<false><<<ROWS, blk, 0, stream>>>(
      ln1f, ff2f, alpha2, beta2, out, nullptr);
}
